// MutualInformationLossFixed_68109591380545
// MI455X (gfx1250) — compile-verified, matrix operand order checked by reading, not run on hardware
//
#include <hip/hip_runtime.h>
#include <hip/hip_bf16.h>

typedef __attribute__((ext_vector_type(16))) _Float16 v16h;
typedef __attribute__((ext_vector_type(8)))  float    v8f;

#define N_SAMPLE   4096
#define IMG_STRIDE 16384  // 1*128*128 elements per batch image

// Workspace layout (float offsets):
//  [0 .. 24)                : stats, 6 vectors * 4 floats {mean, inv_std, sigma, pad}
//  [32 .. 32 + 4*3*4096)    : raw row sums, [pair][pdf:{p1,p2,pj}][4096]
//  [32 + 49152 .. +4)       : per-pair MI partials
#define WS_STATS  0
#define WS_ROWSUM 32
#define WS_MI     (32 + 4 * 3 * N_SAMPLE)

// ---------------------------------------------------------------------------
// Stage 1: per-vector mean, unbiased std (ddof=1), adaptive sigma.
// Vector ids: 0,1 = fused[b]; 2,3 = src1[b]; 4,5 = src2[b].
// ---------------------------------------------------------------------------
__global__ void mi_stats_kernel(const float* __restrict__ fused,
                                const float* __restrict__ src1,
                                const float* __restrict__ src2,
                                float* __restrict__ ws)
{
    const int v = blockIdx.x;  // 0..5
    const float* base = (v < 2) ? fused : (v < 4 ? src1 : src2);
    const float* p = base + (v & 1) * IMG_STRIDE;

    __shared__ float ssum[256];
    __shared__ float ssq[256];
    const int tid = threadIdx.x;

    float s = 0.f, q = 0.f;
    for (int t = tid; t < N_SAMPLE; t += 256) {
        const float x = p[t];
        s += x;
        q += x * x;
    }
    ssum[tid] = s;
    ssq[tid]  = q;
    __syncthreads();
    for (int off = 128; off > 0; off >>= 1) {
        if (tid < off) {
            ssum[tid] += ssum[tid + off];
            ssq[tid]  += ssq[tid + off];
        }
        __syncthreads();
    }
    if (tid == 0) {
        const float mean = ssum[0] * (1.f / (float)N_SAMPLE);
        const float var  = (ssq[0] - (float)N_SAMPLE * mean * mean) *
                           (1.f / (float)(N_SAMPLE - 1));
        const float sd   = sqrtf(fmaxf(var, 0.f));
        const float sig  = fminf(fmaxf(0.2f * sd, 0.001f), 1.0f);
        float* st = ws + WS_STATS + v * 4;
        st[0] = mean;
        st[1] = 1.f / (sd + 1e-8f);   // normalize denominator: std + EPS_STD
        st[2] = sig;
        st[3] = 0.f;
    }
}

// ---------------------------------------------------------------------------
// Stage 2: fused pdf row sums. grid = (64, 4 pairs), block = 128 (4 waves).
// Each wave owns a 16-row strip of the 4096x4096 kernel matrices and sweeps
// j in 32-wide chunks. The three kernel-value tiles (e1, e2, e1*e2) are
// packed as f16 A-matrices and row-reduced with v_wmma_f32_16x16x32_f16
// against an all-ones B-matrix (D = A * ones + C => row sums in every column).
// ---------------------------------------------------------------------------
__global__ void __launch_bounds__(128)
mi_pdf_kernel(const float* __restrict__ fused,
              const float* __restrict__ src1,
              const float* __restrict__ src2,
              float* __restrict__ ws)
{
    const int pair   = blockIdx.y;      // 0..3
    const int b      = pair >> 1;
    const int srcSel = pair & 1;

    const float* x = fused + b * IMG_STRIDE;
    const float* y = (srcSel ? src2 : src1) + b * IMG_STRIDE;
    const float* stx = ws + WS_STATS + b * 4;
    const float* sty = ws + WS_STATS + (2 + srcSel * 2 + b) * 4;

    const float mx = stx[0], isx = stx[1], s1 = stx[2];
    const float my = sty[0], isy = sty[1], s2 = sty[2];
    const float c1 = -0.5f / (s1 * s1);
    const float c2 = -0.5f / (s2 * s2);

    __shared__ float xn[N_SAMPLE];
    __shared__ float yn[N_SAMPLE];
    for (int t = threadIdx.x; t < N_SAMPLE; t += blockDim.x) {
        xn[t] = (x[t] - mx) * isx;
        yn[t] = (y[t] - my) * isy;
    }
    __syncthreads();

    const int wave = threadIdx.x >> 5;
    const int lane = threadIdx.x & 31;
    const int tile = blockIdx.x * 4 + wave;       // 0..255, 16 rows each
    const int row  = tile * 16 + (lane & 15);     // A-matrix M = lane % 16
    const int kbase = (lane < 16) ? 0 : 8;        // A-matrix K group per lane half

    const float xi = xn[row];
    const float yi = yn[row];

    v16h ones;
    #pragma unroll
    for (int e = 0; e < 16; ++e) ones[e] = (_Float16)1.0f;

    v8f acc1 = {}, acc2 = {}, accj = {};

    for (int j0 = 0; j0 < N_SAMPLE; j0 += 32) {
        v16h a1, a2, aj;
        // A 16x32 f16 layout: lane<16 holds K = {0..7, 16..23} + kbase shift,
        // lane>=16 holds K = {8..15, 24..31}; element e pairs map to VGPR e/2.
        #pragma unroll
        for (int g = 0; g < 2; ++g) {
            #pragma unroll
            for (int kk = 0; kk < 8; ++kk) {
                const int j  = j0 + g * 16 + kbase + kk;
                const float d1 = xn[j] - xi;
                const float d2 = yn[j] - yi;
                const float e1 = __expf(c1 * d1 * d1);
                const float e2 = __expf(c2 * d2 * d2);
                const int e = g * 8 + kk;
                a1[e] = (_Float16)e1;
                a2[e] = (_Float16)e2;
                aj[e] = (_Float16)(e1 * e2);
            }
        }
        acc1 = __builtin_amdgcn_wmma_f32_16x16x32_f16(false, a1, false, ones,
                                                      (short)0, acc1, false, false);
        acc2 = __builtin_amdgcn_wmma_f32_16x16x32_f16(false, a2, false, ones,
                                                      (short)0, acc2, false, false);
        accj = __builtin_amdgcn_wmma_f32_16x16x32_f16(false, aj, false, ones,
                                                      (short)0, accj, false, false);
    }

    // C/D layout: lane 0 VGPR r = row r sum; lane 16 VGPR r = row 8+r sum.
    float* r1 = ws + WS_ROWSUM + (pair * 3 + 0) * N_SAMPLE + tile * 16;
    float* r2 = ws + WS_ROWSUM + (pair * 3 + 1) * N_SAMPLE + tile * 16;
    float* rj = ws + WS_ROWSUM + (pair * 3 + 2) * N_SAMPLE + tile * 16;
    if (lane == 0) {
        #pragma unroll
        for (int r = 0; r < 8; ++r) {
            r1[r] = acc1[r];
            r2[r] = acc2[r];
            rj[r] = accj[r];
        }
    } else if (lane == 16) {
        #pragma unroll
        for (int r = 0; r < 8; ++r) {
            r1[8 + r] = acc1[r];
            r2[8 + r] = acc2[r];
            rj[8 + r] = accj[r];
        }
    }
}

// ---------------------------------------------------------------------------
// Stage 3: per-pair MI from raw row sums. One block per pair.
// pdf_raw_i = rowsum_i / N ; p_i = pdf_raw_i / (sum(pdf_raw) + 1e-10)
// MI = sum_i pj_i * log(clip((pj_i+1e-10)/(p1_i*p2_i+1e-10), 1e-10, 1e10))
// ---------------------------------------------------------------------------
__global__ void mi_reduce_kernel(const float* __restrict__ rowsums,
                                 float* __restrict__ mi)
{
    const int pair = blockIdx.x;
    const int tid  = threadIdx.x;
    const float* r1 = rowsums + (pair * 3 + 0) * N_SAMPLE;
    const float* r2 = rowsums + (pair * 3 + 1) * N_SAMPLE;
    const float* rj = rowsums + (pair * 3 + 2) * N_SAMPLE;

    __shared__ float sA[256];
    __shared__ float sB[256];
    __shared__ float sC[256];

    float a = 0.f, b = 0.f, c = 0.f;
    for (int t = tid; t < N_SAMPLE; t += 256) {
        a += r1[t];
        b += r2[t];
        c += rj[t];
    }
    sA[tid] = a; sB[tid] = b; sC[tid] = c;
    __syncthreads();
    for (int off = 128; off > 0; off >>= 1) {
        if (tid < off) {
            sA[tid] += sA[tid + off];
            sB[tid] += sB[tid + off];
            sC[tid] += sC[tid + off];
        }
        __syncthreads();
    }

    const float invN   = 1.f / (float)N_SAMPLE;
    const float inv_d1 = 1.f / (sA[0] * invN + 1e-10f);
    const float inv_d2 = 1.f / (sB[0] * invN + 1e-10f);
    const float inv_dj = 1.f / (sC[0] * invN + 1e-10f);
    __syncthreads();

    float acc = 0.f;
    for (int t = tid; t < N_SAMPLE; t += 256) {
        const float p1 = r1[t] * invN * inv_d1;
        const float p2 = r2[t] * invN * inv_d2;
        const float pj = rj[t] * invN * inv_dj;
        float ratio = (pj + 1e-10f) / (p1 * p2 + 1e-10f);
        ratio = fminf(fmaxf(ratio, 1e-10f), 1e10f);
        acc += pj * __logf(ratio);
    }
    sA[tid] = acc;
    __syncthreads();
    for (int off = 128; off > 0; off >>= 1) {
        if (tid < off) sA[tid] += sA[tid + off];
        __syncthreads();
    }
    if (tid == 0) mi[pair] = sA[0];
}

// ---------------------------------------------------------------------------
// Stage 4: loss = -(sum of 4 MI values) / B, B = 2.
// ---------------------------------------------------------------------------
__global__ void mi_finalize_kernel(const float* __restrict__ mi,
                                   float* __restrict__ out)
{
    out[0] = -(mi[0] + mi[1] + mi[2] + mi[3]) * 0.5f;
}

extern "C" void kernel_launch(void* const* d_in, const int* in_sizes, int n_in,
                              void* d_out, int out_size, void* d_ws, size_t ws_size,
                              hipStream_t stream)
{
    (void)in_sizes; (void)n_in; (void)out_size; (void)ws_size;
    const float* fused = (const float*)d_in[0];
    const float* src1  = (const float*)d_in[1];
    const float* src2  = (const float*)d_in[2];
    float* ws  = (float*)d_ws;
    float* out = (float*)d_out;

    mi_stats_kernel<<<6, 256, 0, stream>>>(fused, src1, src2, ws);
    mi_pdf_kernel<<<dim3(64, 4), 128, 0, stream>>>(fused, src1, src2, ws);
    mi_reduce_kernel<<<4, 256, 0, stream>>>(ws + WS_ROWSUM, ws + WS_MI);
    mi_finalize_kernel<<<1, 1, 0, stream>>>(ws + WS_MI, out);
}
